// ComplexPolarAttention_84121229459551
// MI455X (gfx1250) — matrix operand order repaired and checked by reading, not compiled, not run on hardware
//
#include <hip/hip_runtime.h>
#include <stdint.h>

#define Nn 8192
#define Dd 128
#define Ee 262144
#define EDGE_DIM 50
#define HID4 32

typedef unsigned short ushort_t;
typedef __attribute__((ext_vector_type(16))) __bf16 v16bf;
typedef __attribute__((ext_vector_type(8)))  __bf16 v8bf;
typedef __attribute__((ext_vector_type(8)))  float  v8f;

static __device__ __forceinline__ ushort_t f32_to_bf16(float x) {
    uint32_t u = __float_as_uint(x);
    uint32_t r = u + 0x7FFFu + ((u >> 16) & 1u);   // round-to-nearest-even
    return (ushort_t)(r >> 16);
}

static __device__ __forceinline__ v16bf frag_combine(v8bf lo, v8bf hi) {
    v16bf r;
#pragma unroll
    for (int i = 0; i < 8; ++i) { r[i] = lo[i]; r[8 + i] = hi[i]; }
    return r;
}

// ---------------------------------------------------------------------------
// Kernel 1: build Q (bf16 [N,256] row-major: [cos-part | sin-part]) and
//           Vt (bf16 [256,N]: transposed [mag | phase]) via LDS tile transpose.
// grid (N/32, D/32), block 256
// ---------------------------------------------------------------------------
__global__ void __launch_bounds__(256) prep_kernel(
        const float* __restrict__ mag, const float* __restrict__ phase,
        ushort_t* __restrict__ Q, ushort_t* __restrict__ Vt) {
    __shared__ float lm[32][33];
    __shared__ float lp[32][33];
    int i0 = blockIdx.x * 32;
    int d0 = blockIdx.y * 32;
    int t  = threadIdx.x;
    int ld = t & 31;   // fast dim
    int li = t >> 5;   // 0..7
#pragma unroll
    for (int r = 0; r < 4; ++r) {
        int i = i0 + li + r * 8;
        int d = d0 + ld;
        float mv = mag[(size_t)i * Dd + d];
        float pv = phase[(size_t)i * Dd + d];
        float c = mv * __cosf(pv);
        float s = mv * __sinf(pv);
        Q[(size_t)i * 256 + d]        = f32_to_bf16(c);
        Q[(size_t)i * 256 + 128 + d]  = f32_to_bf16(s);
        lm[li + r * 8][ld] = mv;
        lp[li + r * 8][ld] = pv;
    }
    __syncthreads();
#pragma unroll
    for (int r = 0; r < 4; ++r) {
        int dd = d0 + li + r * 8;
        int ii = i0 + ld;
        float mv = lm[ld][li + r * 8];
        float pv = lp[ld][li + r * 8];
        Vt[(size_t)dd * Nn + ii]          = f32_to_bf16(mv);
        Vt[(size_t)(dd + 128) * Nn + ii]  = f32_to_bf16(pv);
    }
}

// ---------------------------------------------------------------------------
// Kernel 2: per-edge MLP bias = (silu(rbf@W1+b1))@W2 + b2. grid E/256, blk 256
// ---------------------------------------------------------------------------
__global__ void __launch_bounds__(256) edge_bias_kernel(
        const float* __restrict__ rbf,
        const float* __restrict__ W1, const float* __restrict__ b1,
        const float* __restrict__ W2, const float* __restrict__ b2,
        float* __restrict__ bias) {
    int e = blockIdx.x * 256 + threadIdx.x;
    if (e >= Ee) return;
    float r[EDGE_DIM];
    const float* p = rbf + (size_t)e * EDGE_DIM;
#pragma unroll
    for (int k = 0; k < EDGE_DIM; ++k) r[k] = p[k];
    float acc = b2[0];
#pragma unroll 4
    for (int h = 0; h < HID4; ++h) {
        float s = b1[h];
#pragma unroll
        for (int k = 0; k < EDGE_DIM; ++k)
            s = fmaf(r[k], W1[k * HID4 + h], s);       // lane-uniform W1 -> s_loads
        float sig = 1.0f / (1.0f + __expf(-s));
        acc = fmaf(s * sig, W2[h], acc);
    }
    bias[e] = acc;
}

// ---------------------------------------------------------------------------
// Kernel 3: S = Q Q^T / sqrt(128).  4 waves/WG, 64x64 per wave (4x4 frags).
// grid (N/128, N/128), block 128
// ---------------------------------------------------------------------------
__global__ void __launch_bounds__(128) sgemm_kernel(
        const ushort_t* __restrict__ Q, float* __restrict__ S) {
    int lane = threadIdx.x & 31;
    int wave = threadIdx.x >> 5;
    int half = lane >> 4;
    int l16  = lane & 15;
    int rowBase = blockIdx.y * 128 + (wave >> 1) * 64;
    int colBase = blockIdx.x * 128 + (wave & 1) * 64;

    v8f c[4][4] = {};

    for (int kk = 0; kk < 256; kk += 32) {
        v16bf a[4], b[4];
#pragma unroll
        for (int mi = 0; mi < 4; ++mi) {
            // A 16x32 bf16 frag: lane<16 holds K kk..kk+15
            const ushort_t* pa = Q + (size_t)(rowBase + mi * 16 + l16) * 256 + kk + half * 16;
            a[mi] = frag_combine(*(const v8bf*)pa, *(const v8bf*)(pa + 8));
        }
#pragma unroll
        for (int ni = 0; ni < 4; ++ni) {
            // B 32x16 bf16 frag: column n = Q row (colBase+n); lane<16 holds K kk..kk+15
            const ushort_t* pb = Q + (size_t)(colBase + ni * 16 + l16) * 256 + kk + half * 16;
            b[ni] = frag_combine(*(const v8bf*)pb, *(const v8bf*)(pb + 8));
        }
#pragma unroll
        for (int mi = 0; mi < 4; ++mi)
#pragma unroll
            for (int ni = 0; ni < 4; ++ni)
                c[mi][ni] = __builtin_amdgcn_wmma_f32_16x16x32_bf16(
                    false, a[mi], false, b[ni], (short)0, c[mi][ni], false, false);
    }
    const float scale = 0.08838834764831845f;  // 1/sqrt(128)
#pragma unroll
    for (int mi = 0; mi < 4; ++mi)
#pragma unroll
        for (int ni = 0; ni < 4; ++ni)
#pragma unroll
            for (int v = 0; v < 8; ++v) {
                int row = rowBase + mi * 16 + half * 8 + v;
                int col = colBase + ni * 16 + l16;
                S[(size_t)row * Nn + col] = c[mi][ni][v] * scale;
            }
}

// ---------------------------------------------------------------------------
// Kernel 4: scatter edge bias into S with float atomics. grid E/256, blk 256
// ---------------------------------------------------------------------------
__global__ void __launch_bounds__(256) scatter_kernel(
        const long long* __restrict__ ei, const float* __restrict__ bias,
        float* __restrict__ S) {
    int e = blockIdx.x * 256 + threadIdx.x;
    if (e >= Ee) return;
    int i = (int)ei[e];
    int j = (int)ei[(size_t)Ee + e];
    unsafeAtomicAdd(&S[(size_t)i * Nn + j], bias[e]);
}

// ---------------------------------------------------------------------------
// Kernel 5: row softmax; writes bf16 probabilities in place over the f32 row.
// grid N, block 256
// ---------------------------------------------------------------------------
__global__ void __launch_bounds__(256) softmax_kernel(float* __restrict__ S) {
    __shared__ float red[256];
    int row = blockIdx.x;
    int t   = threadIdx.x;
    const float* srow = S + (size_t)row * Nn;
    float vals[32];
    float m = -3.4e38f;
#pragma unroll
    for (int w = 0; w < 32; ++w) {
        vals[w] = srow[t + w * 256];
        m = fmaxf(m, vals[w]);
    }
    red[t] = m;
    __syncthreads();                       // all row reads complete here
    for (int s = 128; s > 0; s >>= 1) {
        if (t < s) red[t] = fmaxf(red[t], red[t + s]);
        __syncthreads();
    }
    m = red[0];
    __syncthreads();
    float sum = 0.0f;
#pragma unroll
    for (int w = 0; w < 32; ++w) {
        vals[w] = __expf(vals[w] - m);
        sum += vals[w];
    }
    red[t] = sum;
    __syncthreads();
    for (int s = 128; s > 0; s >>= 1) {
        if (t < s) red[t] += red[t + s];
        __syncthreads();
    }
    float inv = 1.0f / red[0];
    ushort_t* prow = (ushort_t*)(S + (size_t)row * Nn);  // bf16 in place
#pragma unroll
    for (int w = 0; w < 32; ++w)
        prow[t + w * 256] = f32_to_bf16(vals[w] * inv);
}

// ---------------------------------------------------------------------------
// Kernel 6: O = P @ V. P bf16 rows at byte stride N*4 (in-place), Vt [256,N].
// 4 waves/WG each 32x64 (2x4 frags). grid (256/64, N/128), block 128
// ---------------------------------------------------------------------------
__global__ void __launch_bounds__(128) pv_kernel(
        const float* __restrict__ Sf, const ushort_t* __restrict__ Vt,
        float* __restrict__ out) {
    int lane = threadIdx.x & 31;
    int wave = threadIdx.x >> 5;
    int half = lane >> 4;
    int l16  = lane & 15;
    int rowBase = blockIdx.y * 128 + wave * 32;
    int colBase = blockIdx.x * 64;

    v8f c[2][4] = {};
    const char* Sb = (const char*)Sf;

    for (int kk = 0; kk < Nn; kk += 32) {
        v16bf a[2], b[4];
#pragma unroll
        for (int mi = 0; mi < 2; ++mi) {
            const ushort_t* prow =
                (const ushort_t*)(Sb + (size_t)(rowBase + mi * 16 + l16) * Nn * 4);
            const ushort_t* pa = prow + kk + half * 16;
            a[mi] = frag_combine(*(const v8bf*)pa, *(const v8bf*)(pa + 8));
        }
#pragma unroll
        for (int ni = 0; ni < 4; ++ni) {
            const ushort_t* pb =
                Vt + (size_t)(colBase + ni * 16 + l16) * Nn + kk + half * 16;
            b[ni] = frag_combine(*(const v8bf*)pb, *(const v8bf*)(pb + 8));
        }
#pragma unroll
        for (int mi = 0; mi < 2; ++mi)
#pragma unroll
            for (int ni = 0; ni < 4; ++ni)
                c[mi][ni] = __builtin_amdgcn_wmma_f32_16x16x32_bf16(
                    false, a[mi], false, b[ni], (short)0, c[mi][ni], false, false);
    }
#pragma unroll
    for (int mi = 0; mi < 2; ++mi)
#pragma unroll
        for (int ni = 0; ni < 4; ++ni)
#pragma unroll
            for (int v = 0; v < 8; ++v) {
                int row = rowBase + mi * 16 + half * 8 + v;
                int col = colBase + ni * 16 + l16;
                float val = c[mi][ni][v];
                if (col < 128)
                    out[(size_t)row * 128 + col] = val;                       // new_mag
                else
                    out[(size_t)Nn * 128 + (size_t)row * 128 + (col - 128)] = val; // new_phase
            }
}

// ---------------------------------------------------------------------------
extern "C" void kernel_launch(void* const* d_in, const int* in_sizes, int n_in,
                              void* d_out, int out_size, void* d_ws, size_t ws_size,
                              hipStream_t stream) {
    (void)in_sizes; (void)n_in; (void)out_size; (void)ws_size;
    const float*     mag   = (const float*)d_in[0];
    const float*     phase = (const float*)d_in[1];
    const long long* ei    = (const long long*)d_in[2];
    const float*     rbf   = (const float*)d_in[3];
    const float*     W1    = (const float*)d_in[4];
    const float*     b1    = (const float*)d_in[5];
    const float*     W2    = (const float*)d_in[6];
    const float*     b2    = (const float*)d_in[7];
    float*           out   = (float*)d_out;

    char* ws = (char*)d_ws;
    const size_t S_bytes  = (size_t)Nn * Nn * 4;          // 256 MB
    const size_t Q_bytes  = (size_t)Nn * 256 * 2;         // 4 MB
    float*    S    = (float*)ws;
    ushort_t* Q    = (ushort_t*)(ws + S_bytes);
    ushort_t* Vt   = (ushort_t*)(ws + S_bytes + Q_bytes);
    float*    bias = (float*)(ws + S_bytes + 2 * Q_bytes);

    prep_kernel<<<dim3(Nn / 32, Dd / 32), 256, 0, stream>>>(mag, phase, Q, Vt);
    edge_bias_kernel<<<dim3(Ee / 256), 256, 0, stream>>>(rbf, W1, b1, W2, b2, bias);
    sgemm_kernel<<<dim3(Nn / 128, Nn / 128), 128, 0, stream>>>(Q, S);
    scatter_kernel<<<dim3(Ee / 256), 256, 0, stream>>>(ei, bias, S);
    softmax_kernel<<<dim3(Nn), 256, 0, stream>>>(S);
    pv_kernel<<<dim3(256 / 64, Nn / 128), 128, 0, stream>>>(S, Vt, out);
}
